// Model_28363964023382
// MI455X (gfx1250) — compile-verified, matrix operand order checked by reading, not run on hardware
//
#include <hip/hip_runtime.h>
#include <math.h>

typedef __attribute__((ext_vector_type(16))) _Float16 v16h;
typedef __attribute__((ext_vector_type(8)))  _Float16 v8h;
typedef __attribute__((ext_vector_type(8)))  float    v8f;

#define BN_SCALE 0.99999500003749978f   // 1/sqrt(1+1e-5)

#define NB    64        // batch
#define NODES 512
#define NN    (NB*NODES)   // 32768 nodes
#define NE    262144       // edges
#define GSD   1024
#define MLPH  256
#define PROJ  128
#define HEADS 4
#define CPH   64
#define HC    (HEADS*CPH)  // 256
#define EDIM  32
#define ECATS 6
#define GNNO  256
#define TRUNK 512
#define POLICY 4096
#define NRES  5

// GEMM tiling: 256 threads = 8 waves; block tile 64(M) x 128(N), K-step 32.
// Waves arranged 4(M) x 2(N); each wave computes a 16x64 strip = 4 WMMA tiles.
#define BM 64
#define BN 128
#define BK 32
#define LDSTR 40   // padded LDS stride in halves (80 B: 16B-aligned, conflict-free frags)

// ---------------------------------------------------------------------------
// WMMA fragment layouts per CDNA5 ISA 7.12.2 (wave32):
// A 16x32 f16: lane<16 -> M=lane, elems 0..7 = K 0..7, 8..15 = K 16..23
//              lane>=16 -> M=lane-16, elems 0..7 = K 8..15, 8..15 = K 24..31
// B 32x16 f16: mirrored with N across lanes (read from K-contiguous LDS image)
// C/D 16x16 f32: vgpr r: lane<16 -> M=r, N=lane ; lane>=16 -> M=r+8, N=lane-16
// LDS images are K-contiguous per row/col so each half-fragment is one ds_load_b128.
// ---------------------------------------------------------------------------
__device__ inline v16h frag_from_lds(const _Float16* base, int rowcol, int kb) {
  const v8h* lo = (const v8h*)(base + rowcol * LDSTR + kb);
  const v8h* hi = (const v8h*)(base + rowcol * LDSTR + kb + 16);
  v8h l = *lo, h = *hi;
  v16h f;
#pragma unroll
  for (int j = 0; j < 8; ++j) f[j] = l[j];
#pragma unroll
  for (int j = 0; j < 8; ++j) f[8 + j] = h[j];
  return f;
}

// C = act(A[M,K] @ W[K,N] + bias)
// act: 0 = none ; 1 = relu(bn(x)) ; 2 = relu(bn(x) + resid)
__global__ __launch_bounds__(256) void gemm_wmma_kernel(
    const float* __restrict__ A, const float* __restrict__ W,
    const float* __restrict__ bias, const float* __restrict__ resid,
    float* __restrict__ C, int M, int N, int K, int ldc, int act, float bnscale) {
  __shared__ __align__(16) _Float16 Ah[BM * LDSTR];   //  5120 B
  __shared__ __align__(16) _Float16 Bh[BN * LDSTR];   // 10240 B

  const int lane = threadIdx.x & 31;
  const int wid  = threadIdx.x >> 5;
  const int mBlock = blockIdx.y * BM;
  const int nBlock = blockIdx.x * BN;
  const int wm = wid >> 1;          // 0..3 : wave M row
  const int wn = wid & 1;           // 0..1 : wave N half (64 cols)

  v8f acc[4];
#pragma unroll
  for (int s = 0; s < 4; ++s) acc[s] = (v8f){0.f,0.f,0.f,0.f,0.f,0.f,0.f,0.f};

  // cooperative-load index maps (constant across K loop)
  const int arow = threadIdx.x >> 2;            // 0..63
  const int aseg = (threadIdx.x & 3) * 8;       // 0,8,16,24 : 8 floats/thread
  const int bk   = threadIdx.x >> 3;            // 0..31
  const int bnseg = (threadIdx.x & 7) * 16;     // 16 floats/thread

  for (int kk = 0; kk < K; kk += BK) {
    // ---- stage A tile (BM x BK) as f16, K-contiguous rows
    {
      const float* g = A + (size_t)(mBlock + arow) * K + kk + aseg;
      float4 f0 = *(const float4*)g;
      float4 f1 = *(const float4*)(g + 4);
      v8h pk;
      pk[0] = (_Float16)f0.x; pk[1] = (_Float16)f0.y;
      pk[2] = (_Float16)f0.z; pk[3] = (_Float16)f0.w;
      pk[4] = (_Float16)f1.x; pk[5] = (_Float16)f1.y;
      pk[6] = (_Float16)f1.z; pk[7] = (_Float16)f1.w;
      *(v8h*)(Ah + arow * LDSTR + aseg) = pk;    // ds_store_b128
      if (kk + BK < K)                           // global_prefetch_b8: next A slab
        __builtin_prefetch(g + BK, 0, 0);
    }
    // ---- stage B tile (BK x BN) transposed -> Bh[n][k], K-contiguous cols
    {
      const float* g = W + (size_t)(kk + bk) * N + nBlock + bnseg;
#pragma unroll
      for (int q = 0; q < 4; ++q) {
        float4 f = ((const float4*)g)[q];
        _Float16* d = Bh + (size_t)(bnseg + q * 4) * LDSTR + bk;
        d[0 * LDSTR] = (_Float16)f.x;
        d[1 * LDSTR] = (_Float16)f.y;
        d[2 * LDSTR] = (_Float16)f.z;
        d[3 * LDSTR] = (_Float16)f.w;
      }
    }
    __syncthreads();

    // ---- 4 WMMAs per wave: one A fragment reused across 4 B fragments
    {
      const int afrow = wm * 16 + (lane & 15);
      const int kb = (lane >> 4) << 3;
      v16h a = frag_from_lds(Ah, afrow, kb);
#pragma unroll
      for (int s = 0; s < 4; ++s) {
        const int bcol = wn * 64 + s * 16 + (lane & 15);
        v16h b = frag_from_lds(Bh, bcol, kb);
        acc[s] = __builtin_amdgcn_wmma_f32_16x16x32_f16(
            false, a, false, b, (short)0, acc[s], false, false);
      }
    }
    __syncthreads();
  }

  // ---- epilogue
  const int mbase = mBlock + wm * 16 + ((lane >> 4) << 3);
#pragma unroll
  for (int s = 0; s < 4; ++s) {
    const int n = nBlock + wn * 64 + s * 16 + (lane & 15);
    const float bv = bias ? bias[n] : 0.0f;
#pragma unroll
    for (int r = 0; r < 8; ++r) {
      const int m = mbase + r;
      float v = acc[s][r] + bv;
      if (act == 1) { v *= bnscale; v = v > 0.f ? v : 0.f; }
      else if (act == 2) { v = v * bnscale + resid[(size_t)m * ldc + n]; v = v > 0.f ? v : 0.f; }
      C[(size_t)m * ldc + n] = v;
    }
  }
}

// ---------------------------------------------------------------------------
// Helpers / edge-phase kernels
// ---------------------------------------------------------------------------
__global__ void fill_f32(float* p, float v, int n) {
  int i = blockIdx.x * 256 + threadIdx.x;
  if (i < n) p[i] = v;
}
__global__ void fill_u32(unsigned* p, unsigned v, int n) {
  int i = blockIdx.x * 256 + threadIdx.x;
  if (i < n) p[i] = v;
}

// count incoming edges + sum incoming edge attrs per destination node
__global__ __launch_bounds__(256) void edge_count_attr(
    const int* __restrict__ ei, const int* __restrict__ ecat,
    const float* __restrict__ eemb, float* __restrict__ loop_attr,
    float* __restrict__ cnt) {
  int e = blockIdx.x * 256 + threadIdx.x;
  if (e >= NE) return;
  int d = ei[NE + e];
  int c = ecat[e];
  atomicAdd(&cnt[d], 1.0f);
#pragma unroll 4
  for (int k = 0; k < EDIM; ++k)
    atomicAdd(&loop_attr[(size_t)d * EDIM + k], eemb[c * EDIM + k]);
}

__global__ void loop_attr_div(float* loop_attr, const float* cnt) {
  int i = blockIdx.x * 256 + threadIdx.x;
  if (i >= NN * EDIM) return;
  loop_attr[i] /= fmaxf(cnt[i >> 5], 1.0f);
}

// 6-category edge-embedding transform table (6x256)
__global__ void cat_em_kernel(const float* __restrict__ eemb,
                              const float* __restrict__ We,
                              float* __restrict__ catEm) {
  int i = blockIdx.x * 256 + threadIdx.x;
  if (i >= ECATS * HC) return;
  int c = i >> 8, j = i & 255;
  float s = 0.f;
#pragma unroll 8
  for (int k = 0; k < EDIM; ++k) s += eemb[c * EDIM + k] * We[k * HC + j];
  catEm[i] = s;
}

__device__ inline unsigned enc_f32(float f) {
  unsigned u = __float_as_uint(f);
  return (u >> 31) ? ~u : (u | 0x80000000u);
}
__device__ inline float dec_f32(unsigned u) {
  return (u >> 31) ? __uint_as_float(u ^ 0x80000000u) : __uint_as_float(~u);
}

// pass 1: per (edge, head) attention score + segment max
__global__ __launch_bounds__(256) void gat_scores(
    const int* __restrict__ ei, const int* __restrict__ ecat,
    const float* __restrict__ xl, const float* __restrict__ xr,
    const float* __restrict__ catEm, const float* __restrict__ loop_em,
    const float* __restrict__ att, float* __restrict__ scores,
    unsigned* __restrict__ nmax) {
  int t = blockIdx.x * 256 + threadIdx.x;
  if (t >= (NE + NN) * HEADS) return;
  int e = t >> 2, h = t & 3;
  int s_, d_;
  const float* em;
  if (e < NE) { s_ = ei[e]; d_ = ei[NE + e]; em = catEm + ecat[e] * HC; }
  else        { s_ = d_ = e - NE;            em = loop_em + (size_t)(e - NE) * HC; }
  const float* pl = xl + (size_t)s_ * HC + h * CPH;
  const float* pr = xr + (size_t)d_ * HC + h * CPH;
  const float* pe = em + h * CPH;
  const float* pa = att + h * CPH;
  float sc = 0.f;
#pragma unroll 8
  for (int c = 0; c < CPH; ++c) {
    float v = pl[c] + pr[c] + pe[c];
    v = v > 0.f ? v : 0.2f * v;          // leaky_relu(0.2)
    sc += v * pa[c];
  }
  scores[t] = sc;
  atomicMax(&nmax[d_ * HEADS + h], enc_f32(sc));
}

// pass 2: exp(score - max) + segment sum
__global__ __launch_bounds__(256) void gat_expsum(
    const int* __restrict__ ei, float* __restrict__ scores,
    const unsigned* __restrict__ nmax, float* __restrict__ nsum) {
  int t = blockIdx.x * 256 + threadIdx.x;
  if (t >= (NE + NN) * HEADS) return;
  int e = t >> 2, h = t & 3;
  int d_ = (e < NE) ? ei[NE + e] : (e - NE);
  float m = dec_f32(nmax[d_ * HEADS + h]);
  float ex = __expf(scores[t] - m);
  scores[t] = ex;
  atomicAdd(&nsum[d_ * HEADS + h], ex);
}

// pass 3: alpha-weighted message aggregation into hout
__global__ __launch_bounds__(256) void gat_aggregate(
    const int* __restrict__ ei, const float* __restrict__ scores,
    const float* __restrict__ nsum, const float* __restrict__ xl,
    float* __restrict__ hout) {
  int t = blockIdx.x * 256 + threadIdx.x;
  if (t >= (NE + NN) * HEADS) return;
  int e = t >> 2, h = t & 3;
  int s_, d_;
  if (e < NE) { s_ = ei[e]; d_ = ei[NE + e]; } else { s_ = d_ = e - NE; }
  float alpha = scores[t] / nsum[d_ * HEADS + h];
  const float* pl = xl + (size_t)s_ * HC + h * CPH;
  float* po = hout + (size_t)d_ * HC + h * CPH;
#pragma unroll 4
  for (int c = 0; c < CPH; ++c) atomicAdd(&po[c], pl[c] * alpha);
}

__global__ void node_finalize(float* h, const float* __restrict__ bias, float bnscale) {
  int i = blockIdx.x * 256 + threadIdx.x;
  if (i >= NN * HC) return;
  float v = (h[i] + bias[i & 255]) * bnscale;
  h[i] = v > 0.f ? v : 0.f;
}

// mean over 512 nodes per batch -> zin[:, 256:512]
__global__ void mean_pool(const float* __restrict__ h, float* __restrict__ zin) {
  int b = blockIdx.x, c = threadIdx.x;
  const float* p = h + (size_t)b * NODES * GNNO + c;
  float s = 0.f;
  for (int nd = 0; nd < NODES; ++nd) s += p[(size_t)nd * GNNO];
  zin[b * (MLPH + GNNO) + MLPH + c] = s * (1.0f / NODES);
}

__global__ __launch_bounds__(256) void log_softmax_rows(float* x) {
  int row = blockIdx.x;
  float* p = x + (size_t)row * POLICY;
  __shared__ float red[256];
  float lmax = -INFINITY;
  for (int i = threadIdx.x; i < POLICY; i += 256) lmax = fmaxf(lmax, p[i]);
  red[threadIdx.x] = lmax; __syncthreads();
  for (int s = 128; s > 0; s >>= 1) {
    if ((int)threadIdx.x < s) red[threadIdx.x] = fmaxf(red[threadIdx.x], red[threadIdx.x + s]);
    __syncthreads();
  }
  float m = red[0]; __syncthreads();
  float lsum = 0.f;
  for (int i = threadIdx.x; i < POLICY; i += 256) lsum += __expf(p[i] - m);
  red[threadIdx.x] = lsum; __syncthreads();
  for (int s = 128; s > 0; s >>= 1) {
    if ((int)threadIdx.x < s) red[threadIdx.x] += red[threadIdx.x + s];
    __syncthreads();
  }
  float lse = m + __logf(red[0]);
  for (int i = threadIdx.x; i < POLICY; i += 256) p[i] = p[i] - lse;
}

__global__ void value_head(const float* __restrict__ z, const float* __restrict__ W,
                           const float* __restrict__ b, float* __restrict__ out) {
  int row = threadIdx.x >> 2, col = threadIdx.x & 3;   // 64x4 = 256 threads
  float s = b[col];
  for (int k = 0; k < TRUNK; ++k) s += z[row * TRUNK + k] * W[k * 4 + col];
  out[row * 4 + col] = tanhf(s);
}

// ---------------------------------------------------------------------------
extern "C" void kernel_launch(void* const* d_in, const int* in_sizes, int n_in,
                              void* d_out, int out_size, void* d_ws, size_t ws_size,
                              hipStream_t stream) {
  const float* x_gs  = (const float*)d_in[0];
  const float* x_map = (const float*)d_in[1];
  const int*   ei    = (const int*)d_in[2];
  const int*   ecat  = (const int*)d_in[3];
  const float* gsW1  = (const float*)d_in[4];
  const float* gsb1  = (const float*)d_in[5];
  const float* gsW2  = (const float*)d_in[6];
  const float* gsb2  = (const float*)d_in[7];
  const float* npW   = (const float*)d_in[8];
  const float* npb   = (const float*)d_in[9];
  const float* eemb  = (const float*)d_in[10];
  // gat_params: 3 layers, JAX pytree dict order (sorted keys):
  //   We, Wl, Wr, att, bias, bl, br  -> indices 11..17, 18..24, 25..31
  const float* fin_W = (const float*)d_in[32];
  const float* fin_b = (const float*)d_in[33];
  const float* fus_W = (const float*)d_in[34];
  const float* fus_b = (const float*)d_in[35];
  const float* resW1 = (const float*)d_in[36];
  const float* resb1 = (const float*)d_in[37];
  const float* resW2 = (const float*)d_in[38];
  const float* resb2 = (const float*)d_in[39];
  const float* pol_W = (const float*)d_in[40];
  const float* pol_b = (const float*)d_in[41];
  const float* val_W = (const float*)d_in[42];
  const float* val_b = (const float*)d_in[43];

  // workspace layout (floats)
  float* ws = (float*)d_ws;
  size_t off = 0;
  float* h        = ws + off; off += (size_t)NN * HC;
  float* xl       = ws + off; off += (size_t)NN * HC;
  float* xr       = ws + off; off += (size_t)NN * HC;
  float* loop_em  = ws + off; off += (size_t)NN * HC;
  float* loop_at  = ws + off; off += (size_t)NN * EDIM;
  float* cnt      = ws + off; off += NN;
  float* scores   = ws + off; off += (size_t)(NE + NN) * HEADS;
  unsigned* nmax  = (unsigned*)(ws + off); off += (size_t)NN * HEADS;
  float* nsum     = ws + off; off += (size_t)NN * HEADS;
  float* catEm    = ws + off; off += ECATS * HC;
  float* gs1      = ws + off; off += NB * MLPH;
  float* zin      = ws + off; off += NB * TRUNK;
  float* zA       = ws + off; off += NB * TRUNK;
  float* zB       = ws + off; off += NB * TRUNK;
  float* r1       = ws + off; off += NB * TRUNK;

  auto gemm = [&](const float* A, const float* W, const float* bias,
                  const float* resid, float* C, int M, int Nn, int K, int ldc,
                  int act) {
    dim3 grid(Nn / BN, M / BM);
    gemm_wmma_kernel<<<grid, 256, 0, stream>>>(A, W, bias, resid, C, M, Nn, K,
                                               ldc, act, BN_SCALE);
  };
  auto blocks = [](int n) { return (n + 255) / 256; };

  const int ET = (NE + NN) * HEADS;

  // --- self-loop attr = mean of incoming edge attrs
  fill_f32<<<blocks(NN), 256, 0, stream>>>(cnt, 0.f, NN);
  fill_f32<<<blocks(NN * EDIM), 256, 0, stream>>>(loop_at, 0.f, NN * EDIM);
  edge_count_attr<<<blocks(NE), 256, 0, stream>>>(ei, ecat, eemb, loop_at, cnt);
  loop_attr_div<<<blocks(NN * EDIM), 256, 0, stream>>>(loop_at, cnt);

  // --- game-state MLP (into zin[:, 0:256])
  gemm(x_gs, gsW1, gsb1, nullptr, gs1, NB, MLPH, GSD, MLPH, 1);
  gemm(gs1, gsW2, gsb2, nullptr, zin, NB, MLPH, MLPH, TRUNK, 1);

  // --- node projection: h = relu(bn(x_map @ npW + npb)), packed [NN,128]
  gemm(x_map, npW, npb, nullptr, h, NN, PROJ, 64, PROJ, 1);

  // --- 3 GATv2 layers
  int din = PROJ;
  for (int L = 0; L < 3; ++L) {
    const float* We   = (const float*)d_in[11 + 7 * L + 0];
    const float* Wl   = (const float*)d_in[11 + 7 * L + 1];
    const float* Wr   = (const float*)d_in[11 + 7 * L + 2];
    const float* att  = (const float*)d_in[11 + 7 * L + 3];
    const float* bias = (const float*)d_in[11 + 7 * L + 4];
    const float* bl   = (const float*)d_in[11 + 7 * L + 5];
    const float* br   = (const float*)d_in[11 + 7 * L + 6];

    cat_em_kernel<<<blocks(ECATS * HC), 256, 0, stream>>>(eemb, We, catEm);
    gemm(loop_at, We, nullptr, nullptr, loop_em, NN, HC, EDIM, HC, 0);
    gemm(h, Wl, bl, nullptr, xl, NN, HC, din, HC, 0);
    gemm(h, Wr, br, nullptr, xr, NN, HC, din, HC, 0);

    fill_f32<<<blocks(NN * HC), 256, 0, stream>>>(h, 0.f, NN * HC);
    fill_u32<<<blocks(NN * HEADS), 256, 0, stream>>>(nmax, 0u, NN * HEADS);
    fill_f32<<<blocks(NN * HEADS), 256, 0, stream>>>(nsum, 0.f, NN * HEADS);

    gat_scores<<<blocks(ET), 256, 0, stream>>>(ei, ecat, xl, xr, catEm, loop_em,
                                               att, scores, nmax);
    gat_expsum<<<blocks(ET), 256, 0, stream>>>(ei, scores, nmax, nsum);
    gat_aggregate<<<blocks(ET), 256, 0, stream>>>(ei, scores, nsum, xl, h);
    node_finalize<<<blocks(NN * HC), 256, 0, stream>>>(h, bias, BN_SCALE);
    din = HC;
  }

  // --- final node transform + mean pool (into zin[:, 256:512])
  gemm(h, fin_W, fin_b, nullptr, xl, NN, GNNO, HC, GNNO, 1);
  mean_pool<<<NB, 256, 0, stream>>>(xl, zin);

  // --- fusion trunk + residual blocks
  gemm(zin, fus_W, fus_b, nullptr, zA, NB, TRUNK, TRUNK, TRUNK, 1);
  float* zc = zA;
  float* zn = zB;
  for (int i = 0; i < NRES; ++i) {
    gemm(zc, resW1 + (size_t)i * TRUNK * TRUNK, resb1 + (size_t)i * TRUNK,
         nullptr, r1, NB, TRUNK, TRUNK, TRUNK, 1);
    gemm(r1, resW2 + (size_t)i * TRUNK * TRUNK, resb2 + (size_t)i * TRUNK,
         zc, zn, NB, TRUNK, TRUNK, TRUNK, 2);
    float* t = zc; zc = zn; zn = t;
  }

  // --- heads
  float* pol_out = (float*)d_out;
  gemm(zc, pol_W, pol_b, nullptr, pol_out, NB, POLICY, TRUNK, POLICY, 0);
  log_softmax_rows<<<NB, 256, 0, stream>>>(pol_out);
  value_head<<<1, 256, 0, stream>>>(zc, val_W, val_b, pol_out + (size_t)NB * POLICY);

  (void)in_sizes; (void)n_in; (void)out_size; (void)ws_size;
}
